// DsMiddlePillarEncoder34x8HA_8942121910515
// MI455X (gfx1250) — matrix operand order repaired and checked by reading, not compiled, not run on hardware
//
#include <hip/hip_runtime.h>
#include <hip/hip_bf16.h>
#include <stdint.h>

// ---------- types ----------
typedef __attribute__((ext_vector_type(16))) __bf16 v16bf;
typedef __attribute__((ext_vector_type(8)))  __bf16 v8bf;
typedef __attribute__((ext_vector_type(8)))  float  v8f;

#define C_MID 256
#define GRID94 94
#define GRID47 47
#define NB 2

// per-conv fragment count: 9 taps * 8 cin-blocks * 16 mtiles * 512 bf16
#define FRAG_PER_CONV (9*8*16*512)   // 589824 elems

// ---------- prep: BN fold ----------
__global__ void prep_bn_kernel(const float* __restrict__ g, const float* __restrict__ b,
                               const float* __restrict__ m, const float* __restrict__ v,
                               float* __restrict__ scale, float* __restrict__ bias) {
    int c = threadIdx.x;
    float s = g[c] * rsqrtf(v[c] + 1e-3f);
    scale[c] = s;
    bias[c]  = b[c] - m[c] * s;
}

// ---------- prep: weights -> bf16 WMMA-A fragment layout ----------
// frag[((tap*8+cb)*16+mt)*512 + lane*16 + j]
__global__ __launch_bounds__(256) void prep_w_kernel(const float* __restrict__ w,
                                                     __bf16* __restrict__ frag) {
    long e = (long)blockIdx.x * 256 + threadIdx.x;
    if (e >= (long)FRAG_PER_CONV) return;
    int kx = (int)(e % 3); long t = e / 3;
    int ky = (int)(t % 3); t /= 3;
    int ci = (int)(t % 256);
    int o  = (int)(t / 256);
    int tap = ky*3 + kx;
    int cb  = ci >> 5, kin = ci & 31;
    int mt  = o >> 4,  mm  = o & 15;
    int hi  = kin >> 4;                       // K in {16..31} -> lanes 16..31
    int lane = mm + hi*16;
    int j    = kin & 15;
    frag[(((long)tap*8 + cb)*16 + mt)*512 + lane*16 + j] = (__bf16)w[e];
}

// ---------- zero (16B granularity) ----------
__global__ void zero16_kernel(ulong2* __restrict__ p, long n16) {
    long i = (long)blockIdx.x * 256 + threadIdx.x;
    if (i < n16) p[i] = make_ulong2(0ull, 0ull);
}

// ---------- point MLP + scatter-max ----------
__global__ __launch_bounds__(256) void mlp_scatter_kernel(
    const float* __restrict__ xyz, const int* __restrict__ cnt,
    const float* __restrict__ ptf, const float* __restrict__ mlp_w,
    const float* __restrict__ scale, const float* __restrict__ bias,
    float* __restrict__ grid, int n) {
    __shared__ float Wsh[C_MID*8];
    __shared__ float ssh[C_MID];
    __shared__ float bsh[C_MID];
    int tid = threadIdx.x;
    for (int i = tid; i < C_MID*8; i += 256) Wsh[i] = mlp_w[i];
    ssh[tid] = scale[tid];
    bsh[tid] = bias[tid];
    __syncthreads();
    int lane = tid & 31;
    int pi = blockIdx.x * 8 + (tid >> 5);
    if (pi >= n) return;
    float x = xyz[pi*3+0], y = xyz[pi*3+1], z = xyz[pi*3+2];
    int b = (pi < cnt[0]) ? 0 : 1;
    const float MN = -75.2f, BEV = 1.6f;
    int ix = (int)floorf((x - MN) / BEV); ix = min(max(ix, 0), GRID94-1);
    int iy = (int)floorf((y - MN) / BEV); iy = min(max(iy, 0), GRID94-1);
    float cx = MN + (ix + 0.5f) * BEV;
    float cy = MN + (iy + 0.5f) * BEV;
    float f[8] = {x, y, z, x - cx, y - cy, z, ptf[pi*2+0], ptf[pi*2+1]};
    unsigned* gp = (unsigned*)(grid + (((long)b*GRID94 + iy)*GRID94 + ix)*C_MID);
#pragma unroll
    for (int c8 = 0; c8 < 8; ++c8) {
        int c = lane + c8*32;
        const float* wr = &Wsh[c*8];
        float d = 0.f;
#pragma unroll
        for (int k = 0; k < 8; ++k) d = fmaf(wr[k], f[k], d);
        float h = fmaxf(fmaf(d, ssh[c], bsh[c]), 0.f);
        atomicMax(gp + c, __float_as_uint(h));   // h >= 0 -> bit-monotone
    }
}

// ---------- fp32 grid -> padded bf16 channel-last ----------
__global__ __launch_bounds__(256) void grid_to_pad_kernel(const float* __restrict__ grid,
                                                          __bf16* __restrict__ buf) {
    long i = (long)blockIdx.x * 256 + threadIdx.x;      // one thread = 8 channels
    if (i >= (long)NB*GRID94*GRID94*32) return;
    int c8 = (int)(i & 31); long pix = i >> 5;
    int x = (int)(pix % GRID94); long t = pix / GRID94;
    int y = (int)(t % GRID94);  int b = (int)(t / GRID94);
    const float* src = grid + pix*C_MID + c8*8;
    __bf16* dst = buf + (((long)b*96 + (y+1))*96 + (x+1))*C_MID + c8*8;
    v8bf o;
#pragma unroll
    for (int v = 0; v < 8; ++v) o[v] = (__bf16)src[v];
    *(v8bf*)dst = o;
}

// ---------- 3x3 stride-1 conv, implicit GEMM on WMMA ----------
// workgroup: 8 waves, each wave = 32 couts x 64 pixels (2 Mtiles x 4 Ntiles);
// every A fragment loaded exactly once per workgroup.
template<bool FUSE_ADD>
__global__ __launch_bounds__(256) void conv3x3_s1_kernel(
    const __bf16* __restrict__ inPad, const __bf16* __restrict__ skipPad,
    __bf16* __restrict__ outPad,
    const float* __restrict__ scale, const float* __restrict__ bias,
    const __bf16* __restrict__ wfrag, float* __restrict__ outF32,
    int H, int W) {
    const int HP = H + 2, WP = W + 2;
    const int tilesX = (W + 7) >> 3, tilesY = (H + 7) >> 3;
    int bid = blockIdx.x;
    int tX = bid % tilesX; bid /= tilesX;
    int tY = bid % tilesY; int b = bid / tilesY;
    const int oy0 = tY * 8, ox0 = tX * 8;

    __shared__ __bf16 tile[10*10*C_MID];                // 50 KB

    const int tid = threadIdx.x;
    // stage 10x10 (padded-coord) input patch
    if (oy0 + 10 <= HP && ox0 + 10 <= WP) {
        // fully interior: CDNA5 async copy global->LDS (ASYNCcnt-tracked)
        for (int idx = tid; idx < 10*10*16; idx += 256) {
            int r = idx / 160; int rem = idx - r*160;
            int c = rem >> 4;  int ch = (rem & 15) << 4;
            uint64_t g = (uint64_t)(uintptr_t)(inPad + (((long)b*HP + oy0 + r)*WP + ox0 + c)*C_MID + ch);
            unsigned  l = (unsigned)(uintptr_t)(tile + (r*10 + c)*C_MID + ch);
            asm volatile("global_load_async_to_lds_b128 %0, %1, off"
                         :: "v"(l), "v"(g) : "memory");
        }
        asm volatile("s_wait_asynccnt 0x0" ::: "memory");
    } else {
        // boundary: guarded bounce through VGPRs with zero fill
        for (int idx = tid; idx < 10*10*16; idx += 256) {
            int r = idx / 160; int rem = idx - r*160;
            int c = rem >> 4;  int ch = (rem & 15) << 4;
            int gr = oy0 + r, gc = ox0 + c;
            v16bf val = {};
            if (gr < HP && gc < WP)
                val = *(const v16bf*)(inPad + (((long)b*HP + gr)*WP + gc)*C_MID + ch);
            *(v16bf*)(tile + (r*10 + c)*C_MID + ch) = val;
        }
    }
    __syncthreads();

    const int wave = tid >> 5, lane = tid & 31;
    const int hi = lane >> 4, ln = lane & 15;
    const int mtile0 = wave * 2;                        // 2 mtiles of 16 couts per wave

    int py[4], px[4];
#pragma unroll
    for (int j = 0; j < 4; ++j) {
        int pl = j*16 + ln;
        py[j] = pl >> 3; px[j] = pl & 7;
    }

    v8f acc[2][4] = {};
    const int choff = hi * 16;                          // B lanes: K 0..15 / 16..31

    for (int t = 0; t < 9; ++t) {
        const int ty = t / 3, tx = t - ty*3;
#pragma unroll
        for (int cb = 0; cb < 8; ++cb) {
            v16bf bf[4];
#pragma unroll
            for (int j = 0; j < 4; ++j)
                bf[j] = *(const v16bf*)(tile + ((py[j] + ty)*10 + (px[j] + tx))*C_MID
                                        + cb*32 + choff);
            const __bf16* wk = wfrag + (((long)t*8 + cb)*16)*512 + lane*16;
#pragma unroll
            for (int mt = 0; mt < 2; ++mt) {
                v16bf af = *(const v16bf*)(wk + (long)(mtile0 + mt)*512);
#pragma unroll
                for (int j = 0; j < 4; ++j)
                    acc[mt][j] = __builtin_amdgcn_wmma_f32_16x16x32_bf16(
                        false, af, false, bf[j], (short)0, acc[mt][j], false, false);
            }
        }
    }

#pragma unroll
    for (int j = 0; j < 4; ++j) {
        int oyg = oy0 + py[j], oxg = ox0 + px[j];
        if (oyg >= H || oxg >= W) continue;
        long pixOut = (((long)b*HP + oyg + 1)*WP + oxg + 1)*C_MID;
#pragma unroll
        for (int mt = 0; mt < 2; ++mt) {
            int cbase = wave*32 + mt*16 + hi*8;         // C layout: M = r + 8*hi
            v8bf sk{};
            if (FUSE_ADD) sk = *(const v8bf*)(skipPad + pixOut + cbase);
            v8bf outv;
#pragma unroll
            for (int v = 0; v < 8; ++v) {
                int c = cbase + v;
                float x = fmaf(acc[mt][j][v], scale[c], bias[c]);
                if (FUSE_ADD) x += (float)sk[v];
                x = fmaxf(x, 0.f);
                outv[v] = (__bf16)x;
                if (outF32) outF32[(((long)b*C_MID + c)*H + oyg)*W + oxg] = x;
            }
            *(v8bf*)(outPad + pixOut + cbase) = outv;
        }
    }
}

// ---------- 3x3 stride-2 conv (94 -> 47), B loaded direct from global ----------
__global__ __launch_bounds__(256) void conv3x3_s2_kernel(
    const __bf16* __restrict__ inPad,                   // [2][96][96][256]
    __bf16* __restrict__ outPad,                        // [2][49][49][256]
    const float* __restrict__ scale, const float* __restrict__ bias,
    const __bf16* __restrict__ wfrag) {
    const int H = GRID47, HPo = 49;
    int bid = blockIdx.x;
    int tX = bid % 6; bid /= 6;
    int tY = bid % 6; int b = bid / 6;
    const int oy0 = tY * 8, ox0 = tX * 8;

    const int tid = threadIdx.x;
    const int wave = tid >> 5, lane = tid & 31;
    const int hi = lane >> 4, ln = lane & 15;
    const int mtile0 = wave * 2;

    int oy[4], ox[4], iyb[4], ixb[4];
    bool ok[4];
#pragma unroll
    for (int j = 0; j < 4; ++j) {
        int pl = j*16 + ln;
        oy[j] = oy0 + (pl >> 3); ox[j] = ox0 + (pl & 7);
        ok[j] = (oy[j] < H) && (ox[j] < H);
        iyb[j] = ok[j] ? 2*oy[j] : 0;
        ixb[j] = ok[j] ? 2*ox[j] : 0;
    }

    v8f acc[2][4] = {};
    const int choff = hi * 16;

    for (int t = 0; t < 9; ++t) {
        const int ty = t / 3, tx = t - ty*3;
#pragma unroll
        for (int cb = 0; cb < 8; ++cb) {
            v16bf bf[4];
#pragma unroll
            for (int j = 0; j < 4; ++j)
                bf[j] = *(const v16bf*)(inPad + (((long)b*96 + iyb[j] + ty)*96 + ixb[j] + tx)*C_MID
                                        + cb*32 + choff);
            const __bf16* wk = wfrag + (((long)t*8 + cb)*16)*512 + lane*16;
#pragma unroll
            for (int mt = 0; mt < 2; ++mt) {
                v16bf af = *(const v16bf*)(wk + (long)(mtile0 + mt)*512);
#pragma unroll
                for (int j = 0; j < 4; ++j)
                    acc[mt][j] = __builtin_amdgcn_wmma_f32_16x16x32_bf16(
                        false, af, false, bf[j], (short)0, acc[mt][j], false, false);
            }
        }
    }

#pragma unroll
    for (int j = 0; j < 4; ++j) {
        if (!ok[j]) continue;
        long pixOut = (((long)b*HPo + oy[j] + 1)*HPo + ox[j] + 1)*C_MID;
#pragma unroll
        for (int mt = 0; mt < 2; ++mt) {
            int cbase = wave*32 + mt*16 + hi*8;
            v8bf outv;
#pragma unroll
            for (int v = 0; v < 8; ++v) {
                int c = cbase + v;
                float x = fmaxf(fmaf(acc[mt][j][v], scale[c], bias[c]), 0.f);
                outv[v] = (__bf16)x;
            }
            *(v8bf*)(outPad + pixOut + cbase) = outv;
        }
    }
}

// ---------- host ----------
extern "C" void kernel_launch(void* const* d_in, const int* in_sizes, int n_in,
                              void* d_out, int out_size, void* d_ws, size_t ws_size,
                              hipStream_t stream) {
    (void)n_in; (void)out_size; (void)ws_size;
    const float* xyz   = (const float*)d_in[0];
    const int*   cnt   = (const int*)d_in[1];
    const float* ptf   = (const float*)d_in[2];
    const float* mlp_w = (const float*)d_in[3];
    const int n = in_sizes[0] / 3;

    // input index of each conv's weight (bn g,b,m,v follow at +1..+4)
    const int wIdx[11] = {8, 13, 18, 23, 28, 33, 38, 43, 48, 53, 58};

    // workspace carve-up (all offsets 256B aligned)
    char* ws = (char*)d_ws;
    const long FRAG_BYTES = (long)FRAG_PER_CONV * 2;          // 1,179,648
    __bf16* wfrag = (__bf16*)ws;                              // 11 convs
    long off = 11 * FRAG_BYTES;                               // 12,976,128
    float* bnbuf = (float*)(ws + off); off += 12 * 512 * 4;   // 12 BNs x (scale|bias)
    float* grid  = (float*)(ws + off); off += (long)NB*GRID94*GRID94*C_MID*4;
    const long ABYTES = (long)NB*96*96*C_MID*2;
    const long BBYTES = (long)NB*49*49*C_MID*2;
    __bf16* A0 = (__bf16*)(ws + off); off += ABYTES;
    __bf16* A1 = (__bf16*)(ws + off); off += ABYTES;
    __bf16* A2 = (__bf16*)(ws + off); off += ABYTES;
    __bf16* B0 = (__bf16*)(ws + off); off += BBYTES;
    __bf16* B1 = (__bf16*)(ws + off); off += BBYTES;
    __bf16* B2 = (__bf16*)(ws + off); off += BBYTES;

    // ---- BN folding: bn0 = mlp, bn(1+ci) = conv ci ----
    prep_bn_kernel<<<1, 256, 0, stream>>>((const float*)d_in[4], (const float*)d_in[5],
                                          (const float*)d_in[6], (const float*)d_in[7],
                                          bnbuf, bnbuf + 256);
    for (int ci = 0; ci < 11; ++ci) {
        prep_bn_kernel<<<1, 256, 0, stream>>>(
            (const float*)d_in[wIdx[ci]+1], (const float*)d_in[wIdx[ci]+2],
            (const float*)d_in[wIdx[ci]+3], (const float*)d_in[wIdx[ci]+4],
            bnbuf + (1+ci)*512, bnbuf + (1+ci)*512 + 256);
    }
    // ---- weight repack to WMMA fragments ----
    for (int ci = 0; ci < 11; ++ci) {
        prep_w_kernel<<<(FRAG_PER_CONV + 255)/256, 256, 0, stream>>>(
            (const float*)d_in[wIdx[ci]], wfrag + (long)ci * FRAG_PER_CONV);
    }
    // ---- zero grid + all activation buffers (halos must be 0 each call) ----
    auto zero = [&](void* p, long bytes) {
        long n16 = bytes / 16;
        zero16_kernel<<<(int)((n16 + 255)/256), 256, 0, stream>>>((ulong2*)p, n16);
    };
    zero(grid, (long)NB*GRID94*GRID94*C_MID*4);
    zero(A0, ABYTES); zero(A1, ABYTES); zero(A2, ABYTES);
    zero(B0, BBYTES); zero(B1, BBYTES); zero(B2, BBYTES);

    // ---- points -> dense grid ----
    mlp_scatter_kernel<<<(n + 7)/8, 256, 0, stream>>>(xyz, cnt, ptf, mlp_w,
                                                      bnbuf, bnbuf + 256, grid, n);
    grid_to_pad_kernel<<<(int)(((long)NB*GRID94*GRID94*32 + 255)/256), 256, 0, stream>>>(grid, A0);

    // ---- conv stack ----
    float* x4 = (float*)d_out;                               // [2,256,94,94]
    float* x5 = x4 + (long)NB*C_MID*GRID94*GRID94;           // [2,256,47,47]
    auto S = [&](int ci) { return bnbuf + (1+ci)*512; };
    auto F = [&](int ci) { return wfrag + (long)ci * FRAG_PER_CONV; };
    const int g94 = NB * 12 * 12, g47 = NB * 6 * 6;

    conv3x3_s1_kernel<false><<<g94, 256, 0, stream>>>(A0, nullptr, A1, S(0), S(0)+256, F(0), nullptr, 94, 94);
    conv3x3_s1_kernel<true ><<<g94, 256, 0, stream>>>(A1, A0,      A2, S(1), S(1)+256, F(1), nullptr, 94, 94);
    conv3x3_s1_kernel<false><<<g94, 256, 0, stream>>>(A2, nullptr, A0, S(2), S(2)+256, F(2), nullptr, 94, 94);
    conv3x3_s1_kernel<true ><<<g94, 256, 0, stream>>>(A0, A2,      A1, S(3), S(3)+256, F(3), nullptr, 94, 94);
    conv3x3_s1_kernel<false><<<g94, 256, 0, stream>>>(A1, nullptr, A0, S(4), S(4)+256, F(4), nullptr, 94, 94);
    conv3x3_s1_kernel<true ><<<g94, 256, 0, stream>>>(A0, A1,      A2, S(5), S(5)+256, F(5), x4,      94, 94);
    conv3x3_s2_kernel<<<g47, 256, 0, stream>>>(A2, B0, S(6), S(6)+256, F(6));
    conv3x3_s1_kernel<false><<<g47, 256, 0, stream>>>(B0, nullptr, B1, S(7), S(7)+256, F(7), nullptr, 47, 47);
    conv3x3_s1_kernel<true ><<<g47, 256, 0, stream>>>(B1, B0,      B2, S(8), S(8)+256, F(8), nullptr, 47, 47);
    conv3x3_s1_kernel<false><<<g47, 256, 0, stream>>>(B2, nullptr, B0, S(9), S(9)+256, F(9), nullptr, 47, 47);
    conv3x3_s1_kernel<true ><<<g47, 256, 0, stream>>>(B0, B2,      B1, S(10), S(10)+256, F(10), x5,   47, 47);
}
